// MultiHeadAttention_7576322310204
// MI455X (gfx1250) — matrix operand order repaired and checked by reading, not compiled, not run on hardware
//
#include <hip/hip_runtime.h>
#include <cstdint>

// ---------------- types / helpers ----------------
typedef __bf16 bf16_t;
typedef __attribute__((ext_vector_type(16))) bf16_t v16bf;
typedef __attribute__((ext_vector_type(8)))  bf16_t v8bf;
typedef __attribute__((ext_vector_type(8)))  float  v8f;

static __device__ __forceinline__ v8f wmma_bf16(v16bf a, v16bf b, v8f c) {
  // D = A(16x32 bf16) * B(32x16 bf16) + C(16x16 f32)
  return __builtin_amdgcn_wmma_f32_16x16x32_bf16(
      /*neg_a=*/false, a, /*neg_b=*/false, b,
      /*c_mod=*/(short)0, c, /*reuse_a=*/false, /*reuse_b=*/false);
}

// gfx1250 async global->LDS copy (ASYNCcnt-tracked), guarded by __has_builtin.
// Probe (round 2 stderr) showed the builtin takes generic pointers to
// 'int __attribute__((vector_size(16)))', 4 args total.
#if __has_builtin(__builtin_amdgcn_global_load_async_to_lds_b128)
#define HAVE_ASYNC_LDS 1
typedef int v4i_vs __attribute__((vector_size(16)));
static __device__ __forceinline__ void async_copy_b128(const void* g, void* l) {
  __builtin_amdgcn_global_load_async_to_lds_b128(
      (v4i_vs*)(uintptr_t)g, (v4i_vs*)(uintptr_t)l, 0, 0);
}
static __device__ __forceinline__ void wait_async0() {
#if __has_builtin(__builtin_amdgcn_s_wait_asynccnt)
  __builtin_amdgcn_s_wait_asynccnt(0);
#else
  asm volatile("s_wait_asynccnt 0" ::: "memory");
#endif
}
#endif

// Problem constants (reference: B=2,S=2048,D=1024,H=16,DK=64)
constexpr int BB = 2;
constexpr int SS = 2048;
constexpr int DD = 1024;
constexpr int HH = 16;
constexpr int DKK = 64;
constexpr int MM = BB * SS;     // 4096 rows
constexpr int HD = HH * DKK;    // 1024

// ============================================================
// Kernel 0: weight convert+transpose.  W f32 [K,N] -> Wt bf16 [N,K].
// Puts the K-dim contiguous so WMMA B-fragments are single 32B loads.
// ============================================================
__global__ __launch_bounds__(256) void convert_w_kernel(
    const float* __restrict__ W, bf16_t* __restrict__ Wt, int Kdim, int Ndim) {
  const int idx  = blockIdx.x * 256 + threadIdx.x;  // one (n, k-octet) per thread
  const int octs = Kdim >> 3;
  const int n  = idx / octs;
  const int k0 = (idx - n * octs) << 3;
  if (n >= Ndim) return;
  v8bf o;
#pragma unroll
  for (int e = 0; e < 8; ++e) o[e] = (bf16_t)W[(size_t)(k0 + e) * Ndim + n];
  *(v8bf*)&Wt[(size_t)n * Kdim + k0] = o;
}

// ============================================================
// Kernel 1: QKV projection  (X[M,D] f32) x (Wt[HD,D] bf16) + b -> bf16 [B,H,S,DK]
// 256 threads = 8 waves; wave = 16x64 output strip; block = 32(M) x 256(N)
// ============================================================
__global__ __launch_bounds__(256) void qkv_proj_kernel(
    const float* __restrict__ X, const bf16_t* __restrict__ Wt,
    const float* __restrict__ bias, bf16_t* __restrict__ out, float scale) {
  const int tid   = threadIdx.x;
  const int wave  = tid >> 5;
  const int lane  = tid & 31;
  const int lhalf = lane >> 4;
  const int l15   = lane & 15;
  const int m0 = blockIdx.x * 32 + (wave & 1) * 16;
  const int n0 = blockIdx.y * 256 + (wave >> 1) * 64;

  v8f acc[4];
#pragma unroll
  for (int j = 0; j < 4; ++j)
#pragma unroll
    for (int r = 0; r < 8; ++r) acc[j][r] = 0.0f;

  const float* arow = X + (size_t)(m0 + l15) * DD + lhalf * 16;
  for (int k0 = 0; k0 < DD; k0 += 32) {
    // A fragment: lane row m=l15, K = {16h..16h+15}; f32 -> bf16
    v16bf a;
    {
      const float* ap = arow + k0;
      __builtin_prefetch(ap + 32, 0, 1);
#pragma unroll
      for (int e = 0; e < 8; ++e) a[e] = (bf16_t)ap[e];
#pragma unroll
      for (int e = 0; e < 8; ++e) a[8 + e] = (bf16_t)ap[8 + e];
    }
    // 4 B fragments: contiguous 32B from transposed bf16 weights
#pragma unroll
    for (int j = 0; j < 4; ++j) {
      v16bf bfr = *(const v16bf*)&Wt[(size_t)(n0 + j * 16 + l15) * DD + k0 + lhalf * 16];
      acc[j] = wmma_bf16(a, bfr, acc[j]);
    }
  }

  // epilogue: +bias, *scale, scatter to [B,H,S,DK] bf16
#pragma unroll
  for (int j = 0; j < 4; ++j) {
    const int n  = n0 + j * 16 + l15;
    const float bv = bias[n];
    const int hh = n >> 6, dk = n & 63;
#pragma unroll
    for (int r = 0; r < 8; ++r) {
      const int m = m0 + r + lhalf * 8;
      const int b = m >> 11, s = m & 2047;
      const float v = (acc[j][r] + bv) * scale;
      out[(((size_t)b * HH + hh) * SS + s) * DKK + dk] = (bf16_t)v;
    }
  }
}

// ============================================================
// Kernel 2: flash attention.  Q,K,V bf16 [B,H,S,DK]; out bf16 [B*S, HD]
// One wave per 16-row query tile; 8 waves/block share double-buffered K/V
// chunk tiles in LDS; K staged via async global->LDS (ASYNCcnt) so the next
// chunk's copy overlaps the current chunk's WMMAs + softmax.
// ============================================================
__global__ __launch_bounds__(256) void flash_attn_kernel(
    const bf16_t* __restrict__ Q, const bf16_t* __restrict__ K,
    const bf16_t* __restrict__ V, bf16_t* __restrict__ Aout) {
  __shared__ bf16_t lds_k[2][32][64];    // K chunks, row-major [key][dk]
  __shared__ bf16_t lds_vt[2][64][32];   // V chunks transposed [dk][key]
  __shared__ bf16_t lds_p[8][16][32];    // per-wave P relayout scratch

  const int tid   = threadIdx.x;
  const int wave  = tid >> 5;
  const int lane  = tid & 31;
  const int lhalf = lane >> 4;
  const int l15   = lane & 15;

  const int bh = blockIdx.y;                       // b*H + h
  const int q0 = (blockIdx.x * 8 + wave) * 16;     // query tile base
  const bf16_t* Qb = Q + (size_t)bh * SS * DKK;
  const bf16_t* Kb = K + (size_t)bh * SS * DKK;
  const bf16_t* Vb = V + (size_t)bh * SS * DKK;

  // preload Q fragments (dk 0..31 and 32..63), A-layout
  v16bf qf[2];
#pragma unroll
  for (int f = 0; f < 2; ++f) {
    const bf16_t* qp = Qb + (size_t)(q0 + l15) * DKK + f * 32 + lhalf * 16;
    v8bf lo = *(const v8bf*)qp;
    v8bf hi = *(const v8bf*)(qp + 8);
#pragma unroll
    for (int e = 0; e < 8; ++e) { qf[f][e] = lo[e]; qf[f][8 + e] = hi[e]; }
  }

  v8f o[4];
#pragma unroll
  for (int j = 0; j < 4; ++j)
#pragma unroll
    for (int r = 0; r < 8; ++r) o[j][r] = 0.0f;
  float mrow[8], lrow[8];
#pragma unroll
  for (int r = 0; r < 8; ++r) { mrow[r] = -1e30f; lrow[r] = 0.0f; }

  const int row = tid >> 3;            // 0..31  (cooperative stage index)
  const int col = (tid & 7) * 8;       // 0..56

  // stage one 32-key K/V chunk into LDS buffer `buf`
  auto stage = [&](int buf, int kc) {
    const bf16_t* kg = Kb + (size_t)(kc + row) * DKK + col;
#ifdef HAVE_ASYNC_LDS
    async_copy_b128(kg, &lds_k[buf][row][col]);     // 16B/lane, ASYNCcnt
#else
    *(v8bf*)&lds_k[buf][row][col] = *(const v8bf*)kg;
#endif
    v8bf vv = *(const v8bf*)(Vb + (size_t)(kc + row) * DKK + col);
#pragma unroll
    for (int e = 0; e < 8; ++e) lds_vt[buf][col + e][row] = vv[e];
  };

  // prologue: fill buffer 0
  stage(0, 0);
#ifdef HAVE_ASYNC_LDS
  wait_async0();
#endif
  __syncthreads();

  int cur = 0;
  for (int kc = 0; kc < SS; kc += 32) {
    // issue next chunk's copies into the other buffer (overlaps compute)
    if (kc + 32 < SS) stage(cur ^ 1, kc + 32);

    // ---- scores: 16q x 32k, two 16x16 WMMA tiles, k-dim 64 chained
    v8f sc[2];
#pragma unroll
    for (int j = 0; j < 2; ++j) {
      v8f z;
#pragma unroll
      for (int r = 0; r < 8; ++r) z[r] = 0.0f;
      v16bf kf0 = *(const v16bf*)&lds_k[cur][j * 16 + l15][lhalf * 16];
      v16bf kf1 = *(const v16bf*)&lds_k[cur][j * 16 + l15][32 + lhalf * 16];
      z = wmma_bf16(qf[0], kf0, z);
      sc[j] = wmma_bf16(qf[1], kf1, z);
    }

    // ---- online softmax (rows = r + 8*lhalf; cols spread over 16 lanes)
    float rescale[8];
#pragma unroll
    for (int r = 0; r < 8; ++r) {
      float s0 = sc[0][r], s1 = sc[1][r];
      float mx = fmaxf(s0, s1);
#pragma unroll
      for (int off = 8; off > 0; off >>= 1) mx = fmaxf(mx, __shfl_xor(mx, off));
      const float mnew = fmaxf(mrow[r], mx);
      const float p0 = __expf(s0 - mnew);
      const float p1 = __expf(s1 - mnew);
      rescale[r] = __expf(mrow[r] - mnew);
      float ps = p0 + p1;
#pragma unroll
      for (int off = 8; off > 0; off >>= 1) ps += __shfl_xor(ps, off);
      lrow[r] = lrow[r] * rescale[r] + ps;
      mrow[r] = mnew;
      lds_p[wave][r + lhalf * 8][l15]      = (bf16_t)p0;
      lds_p[wave][r + lhalf * 8][16 + l15] = (bf16_t)p1;
    }
    asm volatile("s_wait_dscnt 0" ::: "memory");  // wave-private LDS RAW

    // ---- reload P as A fragment (16x32 bf16)
    v16bf pf;
    {
      const bf16_t* pp = &lds_p[wave][l15][lhalf * 16];
      v8bf lo = *(const v8bf*)pp;
      v8bf hi = *(const v8bf*)(pp + 8);
#pragma unroll
      for (int e = 0; e < 8; ++e) { pf[e] = lo[e]; pf[8 + e] = hi[e]; }
    }

    // ---- PV: rescale accumulators, 4 WMMAs over DK=64
#pragma unroll
    for (int j = 0; j < 4; ++j) {
      v16bf vf = *(const v16bf*)&lds_vt[cur][j * 16 + l15][lhalf * 16];
      v8f oj = o[j];
#pragma unroll
      for (int r = 0; r < 8; ++r) oj[r] *= rescale[r];
      o[j] = wmma_bf16(pf, vf, oj);
    }

    // next buffer must be fully staged before anyone reads it / rewrites cur
#ifdef HAVE_ASYNC_LDS
    wait_async0();
#endif
    __syncthreads();
    cur ^= 1;
  }

  // ---- normalize and store attn output as bf16 [B*S, HD]
  const int b = bh >> 4, hh = bh & 15;
  float inv[8];
#pragma unroll
  for (int r = 0; r < 8; ++r) inv[r] = 1.0f / lrow[r];
#pragma unroll
  for (int j = 0; j < 4; ++j) {
#pragma unroll
    for (int r = 0; r < 8; ++r) {
      const int srow = q0 + r + lhalf * 8;
      Aout[(size_t)(b * SS + srow) * HD + hh * 64 + j * 16 + l15] =
          (bf16_t)(o[j][r] * inv[r]);
    }
  }
}

// ============================================================
// Kernel 3: output projection  (A[M,HD] bf16) x (Wot[D,HD] bf16) + bo -> f32 [M,D]
// ============================================================
__global__ __launch_bounds__(256) void out_proj_kernel(
    const bf16_t* __restrict__ A, const bf16_t* __restrict__ Wt,
    const float* __restrict__ bias, float* __restrict__ out) {
  const int tid   = threadIdx.x;
  const int wave  = tid >> 5;
  const int lane  = tid & 31;
  const int lhalf = lane >> 4;
  const int l15   = lane & 15;
  const int m0 = blockIdx.x * 32 + (wave & 1) * 16;
  const int n0 = blockIdx.y * 256 + (wave >> 1) * 64;

  v8f acc[4];
#pragma unroll
  for (int j = 0; j < 4; ++j)
#pragma unroll
    for (int r = 0; r < 8; ++r) acc[j][r] = 0.0f;

  const bf16_t* arow = A + (size_t)(m0 + l15) * HD + lhalf * 16;
  for (int k0 = 0; k0 < HD; k0 += 32) {
    v16bf a;
    {
      const bf16_t* ap = arow + k0;
      v8bf lo = *(const v8bf*)ap;
      v8bf hi = *(const v8bf*)(ap + 8);
#pragma unroll
      for (int e = 0; e < 8; ++e) { a[e] = lo[e]; a[8 + e] = hi[e]; }
    }
#pragma unroll
    for (int j = 0; j < 4; ++j) {
      v16bf bfr = *(const v16bf*)&Wt[(size_t)(n0 + j * 16 + l15) * HD + k0 + lhalf * 16];
      acc[j] = wmma_bf16(a, bfr, acc[j]);
    }
  }

#pragma unroll
  for (int j = 0; j < 4; ++j) {
    const int n = n0 + j * 16 + l15;
    const float bv = bias[n];
#pragma unroll
    for (int r = 0; r < 8; ++r) {
      const int m = m0 + r + lhalf * 8;
      out[(size_t)m * DD + n] = acc[j][r] + bv;
    }
  }
}

// ============================================================
extern "C" void kernel_launch(void* const* d_in, const int* in_sizes, int n_in,
                              void* d_out, int out_size, void* d_ws, size_t ws_size,
                              hipStream_t stream) {
  const float* query = (const float*)d_in[0];
  const float* key   = (const float*)d_in[1];
  const float* value = (const float*)d_in[2];
  const float* Wq    = (const float*)d_in[3];
  const float* bq    = (const float*)d_in[4];
  const float* Wk    = (const float*)d_in[5];
  const float* bk    = (const float*)d_in[6];
  const float* Wv    = (const float*)d_in[7];
  const float* bv    = (const float*)d_in[8];
  const float* Wo    = (const float*)d_in[9];
  const float* bo    = (const float*)d_in[10];
  float* out = (float*)d_out;

  bf16_t* ws = (bf16_t*)d_ws;
  const size_t NQ = (size_t)BB * HH * SS * DKK;  // 4 Mi elements
  const size_t NW = (size_t)DD * HD;             // 1 Mi elements per weight
  bf16_t* Qb  = ws;
  bf16_t* Kb  = ws + NQ;
  bf16_t* Vb  = ws + 2 * NQ;
  bf16_t* Ab  = ws + 3 * NQ;
  bf16_t* Wqt = ws + 4 * NQ;            // [HD, D] bf16 (K contiguous)
  bf16_t* Wkt = Wqt + NW;
  bf16_t* Wvt = Wkt + NW;
  bf16_t* Wot = Wvt + NW;               // [D, HD] bf16

  const dim3 blk(256);

  // one-time weight convert+transpose (K-dim made contiguous for B-fragments)
  const int cgrid = (int)(NW / 8 / 256);  // 512 blocks
  convert_w_kernel<<<cgrid, blk, 0, stream>>>(Wq, Wqt, DD, HD);
  convert_w_kernel<<<cgrid, blk, 0, stream>>>(Wk, Wkt, DD, HD);
  convert_w_kernel<<<cgrid, blk, 0, stream>>>(Wv, Wvt, DD, HD);
  convert_w_kernel<<<cgrid, blk, 0, stream>>>(Wo, Wot, HD, DD);

  const dim3 gp(MM / 32, HD / 256);
  qkv_proj_kernel<<<gp, blk, 0, stream>>>(query, Wqt, bq, Qb, 0.125f);  // 1/sqrt(64)
  qkv_proj_kernel<<<gp, blk, 0, stream>>>(key,   Wkt, bk, Kb, 1.0f);
  qkv_proj_kernel<<<gp, blk, 0, stream>>>(value, Wvt, bv, Vb, 1.0f);

  flash_attn_kernel<<<dim3(SS / 128, BB * HH), blk, 0, stream>>>(Qb, Kb, Vb, Ab);

  out_proj_kernel<<<dim3(MM / 32, DD / 256), blk, 0, stream>>>(Ab, Wot, bo, out);
}
